// SmallerNN_62036507623874
// MI455X (gfx1250) — 8-bit matrix operand order repaired and checked by reading, compiled, not run on hardware
//
#include <hip/hip_runtime.h>
#include <hip/hip_bf16.h>

typedef __attribute__((ext_vector_type(16))) int   v16i;
typedef __attribute__((ext_vector_type(8)))  float v8f;
typedef __attribute__((ext_vector_type(2)))  int   v2i;
typedef __attribute__((ext_vector_type(4)))  int   v4i;

#define FP8_P1 0x38u
#define FP8_M1 0xB8u

#if __has_builtin(__builtin_amdgcn_global_load_async_to_lds_b128)
#define HAS_ASYNC_LDS 1
#else
#define HAS_ASYNC_LDS 0
#endif

typedef __attribute__((address_space(1))) v4i gas_v4i;
typedef __attribute__((address_space(3))) v4i lds_v4i;

__device__ __forceinline__ void copy16_to_lds(const unsigned char* g, unsigned char* l) {
#if HAS_ASYNC_LDS
  __builtin_amdgcn_global_load_async_to_lds_b128(
      (gas_v4i*)(size_t)g, (lds_v4i*)(unsigned)(size_t)l, 0, 0);
#else
  *(v4i*)l = *(const v4i*)g;
#endif
}

__device__ __forceinline__ void wait_async_le2() {
#if HAS_ASYNC_LDS
#if __has_builtin(__builtin_amdgcn_s_wait_asynccnt)
  __builtin_amdgcn_s_wait_asynccnt(2);
#else
  asm volatile("s_wait_asynccnt 0x2" ::: "memory");
#endif
#endif
}

__device__ __forceinline__ void wait_async_le0() {
#if HAS_ASYNC_LDS
#if __has_builtin(__builtin_amdgcn_s_wait_asynccnt)
  __builtin_amdgcn_s_wait_asynccnt(0);
#else
  asm volatile("s_wait_asynccnt 0x0" ::: "memory");
#endif
#endif
}

__device__ __forceinline__ unsigned pack4(unsigned b0, unsigned b1, unsigned b2, unsigned b3) {
  return b0 | (b1 << 8) | (b2 << 16) | (b3 << 24);
}

__global__ void bnn_quant_input(const float* __restrict__ x, unsigned* __restrict__ out, int n4) {
  int i = blockIdx.x * blockDim.x + threadIdx.x;
  if (i >= n4) return;
  float4 v = ((const float4*)x)[i];
  out[i] = pack4(v.x >= 0.5f ? FP8_P1 : FP8_M1,
                 v.y >= 0.5f ? FP8_P1 : FP8_M1,
                 v.z >= 0.5f ? FP8_P1 : FP8_M1,
                 v.w >= 0.5f ? FP8_P1 : FP8_M1);
}

__global__ void bnn_quant_weight(const float* __restrict__ w, unsigned* __restrict__ out, int n4) {
  int i = blockIdx.x * blockDim.x + threadIdx.x;
  if (i >= n4) return;
  float4 v = ((const float4*)w)[i];
  out[i] = pack4(v.x >= 0.0f ? FP8_P1 : FP8_M1,
                 v.y >= 0.0f ? FP8_P1 : FP8_M1,
                 v.z >= 0.0f ? FP8_P1 : FP8_M1,
                 v.w >= 0.0f ? FP8_P1 : FP8_M1);
}

__device__ __forceinline__ v16i load_afrag(const unsigned char* arow_kb, int hi) {
  v16i a;
#pragma unroll
  for (int j = 0; j < 8; ++j) {
    const int off = 32 * (j >> 1) + 8 * (j & 1) + 16 * hi;
    v2i d = *(const v2i*)(arow_kb + off);
    a[2 * j]     = d[0];
    a[2 * j + 1] = d[1];
  }
  return a;
}

__global__ __launch_bounds__(256)
void bnn_gemm_bn_binq(const unsigned char* __restrict__ A,
                      const unsigned char* __restrict__ Wq,
                      const float* __restrict__ gn, const float* __restrict__ bn_,
                      const float* __restrict__ mn, const float* __restrict__ vn,
                      unsigned char* __restrict__ out,
                      int M, int N, int K)
{
  __shared__ __align__(16) unsigned char smem[2][64 * 128];

  const int tid   = threadIdx.x;
  const int lane  = tid & 31;
  const int wid   = tid >> 5;
  const int lo    = lane & 15;
  const int hi    = lane >> 4;
  const int mbase = (blockIdx.y * 8 + wid) * 32;
  const int nbase = blockIdx.x * 64;

  const int e0 = tid * 16;
  const int e1 = 4096 + tid * 16;
  const unsigned char* bsrc0 = Wq + (size_t)(nbase + (e0 >> 7)) * (size_t)K + (e0 & 127);
  const unsigned char* bsrc1 = Wq + (size_t)(nbase + (e1 >> 7)) * (size_t)K + (e1 & 127);

  const unsigned char* arow0 = A + (size_t)(mbase + lo) * (size_t)K;
  const unsigned char* arow1 = arow0 + (size_t)16 * (size_t)K;

  v8f acc[8] = {};

  const int nsteps = K >> 7;

  copy16_to_lds(bsrc0, &smem[0][e0]);
  copy16_to_lds(bsrc1, &smem[0][e1]);

  for (int s = 0; s < nsteps; ++s) {
    const int kb  = s << 7;
    const int cur = s & 1;
    const bool has_next = (s + 1 < nsteps);

    if (has_next) {
      copy16_to_lds(bsrc0 + kb + 128, &smem[cur ^ 1][e0]);
      copy16_to_lds(bsrc1 + kb + 128, &smem[cur ^ 1][e1]);
      wait_async_le2();
      __builtin_prefetch(arow0 + kb + 128, 0, 0);
      __builtin_prefetch(arow1 + kb + 128, 0, 0);
    } else {
      wait_async_le0();
    }
    __syncthreads();

    const v16i a0 = load_afrag(arow0 + kb, hi);
    const v16i a1 = load_afrag(arow1 + kb, hi);

#pragma unroll
    for (int t = 0; t < 4; ++t) {
      v16i bfrag;
#pragma unroll
      for (int q = 0; q < 4; ++q) {
        v4i d = *(const v4i*)(&smem[cur][(16 * t + lo) * 128 + 32 * q + 16 * hi]);
        bfrag[4 * q]     = d[0];
        bfrag[4 * q + 1] = d[1];
        bfrag[4 * q + 2] = d[2];
        bfrag[4 * q + 3] = d[3];
      }
      acc[t]     = __builtin_amdgcn_wmma_f32_16x16x128_fp8_fp8(
                       a0, bfrag, (short)0, acc[t],     false, false);
      acc[4 + t] = __builtin_amdgcn_wmma_f32_16x16x128_fp8_fp8(
                       a1, bfrag, (short)0, acc[4 + t], false, false);
    }
    __syncthreads();
  }

#pragma unroll
  for (int t = 0; t < 4; ++t) {
    const int n = nbase + 16 * t + lo;
    const float inv = gn[n] * rsqrtf(vn[n] + 1e-5f);
    const float mu  = mn[n];
    const float bb  = bn_[n];
#pragma unroll
    for (int half = 0; half < 2; ++half) {
      const v8f a = acc[half * 4 + t];
#pragma unroll
      for (int r = 0; r < 8; ++r) {
        const float y = (a[r] - mu) * inv + bb;
        const int mrow = mbase + 16 * half + r + 8 * hi;
        out[(size_t)mrow * (size_t)N + (size_t)n] =
            (y >= 0.0f) ? (unsigned char)FP8_P1 : (unsigned char)FP8_M1;
      }
    }
  }
}

__global__ void bnn_final(const unsigned char* __restrict__ A,
                          const unsigned char* __restrict__ W,
                          float* __restrict__ out, int M, int K)
{
  int m = blockIdx.x * blockDim.x + threadIdx.x;
  if (m >= M) return;
  const uint4* a  = (const uint4*)(A + (size_t)m * (size_t)K);
  const uint4* w0 = (const uint4*)(W);
  const uint4* w1 = (const uint4*)(W + K);
  int neg0 = 0, neg1 = 0;
  for (int i = 0; i < K / 16; ++i) {
    uint4 av = a[i], b0 = w0[i], b1 = w1[i];
    neg0 += __popc((av.x ^ b0.x) & 0x80808080u) + __popc((av.y ^ b0.y) & 0x80808080u)
          + __popc((av.z ^ b0.z) & 0x80808080u) + __popc((av.w ^ b0.w) & 0x80808080u);
    neg1 += __popc((av.x ^ b1.x) & 0x80808080u) + __popc((av.y ^ b1.y) & 0x80808080u)
          + __popc((av.z ^ b1.z) & 0x80808080u) + __popc((av.w ^ b1.w) & 0x80808080u);
  }
  out[2 * m + 0] = (float)(K - 2 * neg0);
  out[2 * m + 1] = (float)(K - 2 * neg1);
}

extern "C" void kernel_launch(void* const* d_in, const int* in_sizes, int n_in,
                              void* d_out, int out_size, void* d_ws, size_t ws_size,
                              hipStream_t stream) {
  (void)in_sizes; (void)n_in; (void)out_size; (void)ws_size;
  const int B = 16384, D = 1024, H = 2048;

  const float* x    = (const float*)d_in[0];
  const float* w1   = (const float*)d_in[1];
  const float* g1   = (const float*)d_in[2];
  const float* b1   = (const float*)d_in[3];
  const float* m1   = (const float*)d_in[4];
  const float* v1   = (const float*)d_in[5];
  const float* w2   = (const float*)d_in[6];
  const float* g2   = (const float*)d_in[7];
  const float* b2   = (const float*)d_in[8];
  const float* m2   = (const float*)d_in[9];
  const float* v2   = (const float*)d_in[10];
  const float* w3   = (const float*)d_in[11];
  const float* g3   = (const float*)d_in[12];
  const float* b3   = (const float*)d_in[13];
  const float* m3   = (const float*)d_in[14];
  const float* v3   = (const float*)d_in[15];
  const float* wout = (const float*)d_in[16];

  char* ws = (char*)d_ws;
  const size_t MB = 1u << 20;
  unsigned char* A0  = (unsigned char*)(ws + 0);
  unsigned char* A1  = (unsigned char*)(ws + 16 * MB);
  unsigned char* A2  = (unsigned char*)(ws + 48 * MB);
  unsigned char* A3  = (unsigned char*)(ws + 0);
  unsigned char* W1q = (unsigned char*)(ws + 80 * MB);
  unsigned char* W2q = (unsigned char*)(ws + 82 * MB);
  unsigned char* W3q = (unsigned char*)(ws + 86 * MB);
  unsigned char* Woq = (unsigned char*)(ws + 90 * MB);

  { int n4 = B * D / 4; bnn_quant_input <<<(n4 + 255) / 256, 256, 0, stream>>>(x,    (unsigned*)A0,  n4); }
  { int n4 = H * D / 4; bnn_quant_weight<<<(n4 + 255) / 256, 256, 0, stream>>>(w1,   (unsigned*)W1q, n4); }
  { int n4 = H * H / 4; bnn_quant_weight<<<(n4 + 255) / 256, 256, 0, stream>>>(w2,   (unsigned*)W2q, n4); }
  { int n4 = H * H / 4; bnn_quant_weight<<<(n4 + 255) / 256, 256, 0, stream>>>(w3,   (unsigned*)W3q, n4); }
  { int n4 = 2 * H / 4; bnn_quant_weight<<<(n4 + 255) / 256, 256, 0, stream>>>(wout, (unsigned*)Woq, n4); }

  dim3 blk(256);
  dim3 grd(H / 64, B / 256);
  bnn_gemm_bn_binq<<<grd, blk, 0, stream>>>(A0, W1q, g1, b1, m1, v1, A1, B, H, D);
  bnn_gemm_bn_binq<<<grd, blk, 0, stream>>>(A1, W2q, g2, b2, m2, v2, A2, B, H, H);
  bnn_gemm_bn_binq<<<grd, blk, 0, stream>>>(A2, W3q, g3, b3, m3, v3, A3, B, H, H);

  bnn_final<<<B / 256, 256, 0, stream>>>(A3, Woq, (float*)d_out, B, H);
}
